// MBSClassifier_6390911337225
// MI455X (gfx1250) — matrix operand order repaired and checked by reading, not compiled, not run on hardware
//
#include <hip/hip_runtime.h>
#include <hip/hip_bf16.h>
#include <math.h>

typedef __attribute__((ext_vector_type(16))) _Float16 v16h;
typedef __attribute__((ext_vector_type(8)))  _Float16 v8h;
typedef __attribute__((ext_vector_type(4)))  _Float16 v4h;
typedef __attribute__((ext_vector_type(8)))  float    v8f;
typedef __attribute__((ext_vector_type(4)))  float    v4f;

#define N      8192
#define NFEAT  128

// ---------------------------------------------------------------------------
// Kernel 1: per-row degree count (adj > 0) fused with f32 -> f16 convert.
// The 268 MB f32 stream is single-use -> non-temporal loads (TH=NT) so it
// does not evict the f16 copy (134 MB) from the 192 MB L2; adj16 writes use
// default RT policy so both WMMA passes can hit L2 instead of HBM.
// ---------------------------------------------------------------------------
__global__ __launch_bounds__(256) void deg_convert_kernel(
    const float* __restrict__ adj, _Float16* __restrict__ adj16,
    float* __restrict__ dinv)
{
    __shared__ float scnt[256];
    const int row = blockIdx.x;
    const v4f* src = (const v4f*)(adj + (size_t)row * N);
    v4h* dst = (v4h*)(adj16 + (size_t)row * N);
    int cnt = 0;
    for (int j4 = threadIdx.x; j4 < N / 4; j4 += 256) {
        v4f a = __builtin_nontemporal_load(src + j4);   // global_load_b128 th:NT
        cnt += (a.x > 0.0f) + (a.y > 0.0f) + (a.z > 0.0f) + (a.w > 0.0f);
        v4h h = { (_Float16)a.x, (_Float16)a.y, (_Float16)a.z, (_Float16)a.w };
        dst[j4] = h;
    }
    scnt[threadIdx.x] = (float)cnt;
    __syncthreads();
    for (int s = 128; s > 0; s >>= 1) {
        if (threadIdx.x < s) scnt[threadIdx.x] += scnt[threadIdx.x + s];
        __syncthreads();
    }
    if (threadIdx.x == 0) dinv[row] = rsqrtf(scnt[0]);
}

// ---------------------------------------------------------------------------
// Kernel 2: s1T[c][i] = dinv[i] * (x[i,:] @ W1[:,c]),  c<10, padded to 16.
// Stored transposed in f16 so the WMMA B fragment is a contiguous load.
// ---------------------------------------------------------------------------
__global__ __launch_bounds__(256) void s1_kernel(
    const float* __restrict__ x, const float* __restrict__ W1,
    const float* __restrict__ dinv, _Float16* __restrict__ s1T)
{
    const int t = blockIdx.x * 256 + threadIdx.x;   // N*16 threads
    const int c = t & 15;
    const int i = t >> 4;
    float acc = 0.0f;
    if (c < 10) {
        const float* xr = x + (size_t)i * NFEAT;
        #pragma unroll 4
        for (int f = 0; f < NFEAT; ++f) acc += xr[f] * W1[f * 10 + c];
        acc *= dinv[i];
    }
    s1T[(size_t)c * N + i] = (_Float16)acc;
}

// ---------------------------------------------------------------------------
// Kernel 3/5: WMMA pass.  t = adj16 @ s  (16-col padded, transposed B).
// Block = 256 threads = 8 waves; one 16-row output tile per block; each wave
// covers a K-segment of 1024 (32 steps of v_wmma_f32_16x16x32_f16).
// Partials combined via LDS ds_add_f32; wave 0 runs the epilogue.
//   PASS==1: hT[n][i] = relu(dinv*(t + s1) + b1)        (n<10, else 0)
//   PASS==2: colsum[n] += dinv*(t + s2) + b2            (n<5)
// ---------------------------------------------------------------------------
template <int PASS>
__global__ __launch_bounds__(256) void gcn_pass_kernel(
    const _Float16* __restrict__ adj16, const _Float16* __restrict__ sT,
    const float* __restrict__ dinv, const float* __restrict__ bias,
    float* __restrict__ hT, float* __restrict__ colsum)
{
    __shared__ float acc_lds[256];                 // 32 lanes x 8 accum regs
    const int tid  = threadIdx.x;
    const int lane = tid & 31;
    const int wave = tid >> 5;
    const int g    = lane >> 4;                    // lane group (0/1)
    const int mn   = lane & 15;                    // A row / B,C column
    const int i0   = blockIdx.x * 16;              // tile row base

    acc_lds[tid] = 0.0f;
    __syncthreads();

    // A: lane streams its adj16 row; per K=32 step: halves [k0+8g..+7] and
    //    [k0+16+8g..+7]  (ISA 16-bit A 16x32 layout) -> two b128 loads.
    const _Float16* arow = adj16 + (size_t)(i0 + mn) * N + 8 * g;
    // B: lane reads column mn of s (row mn of sT); K halves [k0+8g..+7] and [k0+16+8g..+7].
    const _Float16* brow = sT + (size_t)mn * N + 8 * g;

    v8f acc = {};
    const int kbeg = wave * (N / 8);
    const int kend = kbeg + (N / 8);
    for (int k0 = kbeg; k0 < kend; k0 += 32) {
        v8h alo = *(const v8h*)(arow + k0);
        v8h ahi = *(const v8h*)(arow + k0 + 16);
        v8h blo = *(const v8h*)(brow + k0);
        v8h bhi = *(const v8h*)(brow + k0 + 16);
        v16h a = __builtin_shufflevector(alo, ahi, 0,1,2,3,4,5,6,7,8,9,10,11,12,13,14,15);
        v16h b = __builtin_shufflevector(blo, bhi, 0,1,2,3,4,5,6,7,8,9,10,11,12,13,14,15);
        acc = __builtin_amdgcn_wmma_f32_16x16x32_f16(
            /*neg_a=*/false, a, /*neg_b=*/false, b,
            /*c_mod=*/(short)0, acc, /*reuse_a=*/false, /*reuse_b=*/false);
    }

    #pragma unroll
    for (int v = 0; v < 8; ++v)
        atomicAdd(&acc_lds[lane * 8 + v], acc[v]);   // ds_add_f32
    __syncthreads();

    if (wave == 0) {
        const int n     = mn;                       // output column
        const int ibase = i0 + 8 * g;               // first of this lane's 8 rows
        v8h sv = *(const v8h*)(sT + (size_t)n * N + ibase);   // identity term
        const int ncols = (PASS == 1) ? 10 : 5;
        const float bn  = (n < ncols) ? bias[n] : 0.0f;
        float psum = 0.0f;
        #pragma unroll
        for (int v = 0; v < 8; ++v) {
            const int i = ibase + v;
            float o = dinv[i] * (acc_lds[lane * 8 + v] + (float)sv[v]) + bn;
            if (PASS == 1) {
                hT[(size_t)n * N + i] = (n < 10) ? fmaxf(o, 0.0f) : 0.0f;
            } else {
                if (n < 5) psum += o;
            }
        }
        if (PASS == 2 && n < 5) atomicAdd(&colsum[n], psum);
    }
}

// ---------------------------------------------------------------------------
// Kernel 4: s2T[c][i] = dinv[i] * (h[i,:] @ W2[:,c]), c<5, padded to 16.
// Also zeroes the colsum accumulator (runs before pass 2 in stream order).
// ---------------------------------------------------------------------------
__global__ __launch_bounds__(256) void s2_kernel(
    const float* __restrict__ hT, const float* __restrict__ W2,
    const float* __restrict__ dinv, _Float16* __restrict__ s2T,
    float* __restrict__ colsum)
{
    const int t = blockIdx.x * 256 + threadIdx.x;   // N*16 threads
    const int c = t & 15;
    const int i = t >> 4;
    float acc = 0.0f;
    if (c < 5) {
        #pragma unroll
        for (int k = 0; k < 10; ++k) acc += hT[(size_t)k * N + i] * W2[k * 5 + c];
        acc *= dinv[i];
    }
    s2T[(size_t)c * N + i] = (_Float16)acc;
    if (t < 8) colsum[t] = 0.0f;
}

// ---------------------------------------------------------------------------
// Kernel 6: head.  z = relu(mean); z2 = relu(z@fc1+b); y = sigmoid(z2@fc+b).
// ---------------------------------------------------------------------------
__global__ void finalize_kernel(
    const float* __restrict__ colsum,
    const float* __restrict__ fc1_W, const float* __restrict__ fc1_b,
    const float* __restrict__ fc_W,  const float* __restrict__ fc_b,
    float* __restrict__ out)
{
    if (threadIdx.x == 0 && blockIdx.x == 0) {
        float z[5], z2[5];
        for (int c = 0; c < 5; ++c) z[c] = fmaxf(colsum[c] * (1.0f / (float)N), 0.0f);
        for (int j = 0; j < 5; ++j) {
            float a = fc1_b[j];
            for (int k = 0; k < 5; ++k) a += z[k] * fc1_W[k * 5 + j];
            z2[j] = fmaxf(a, 0.0f);
        }
        for (int c = 0; c < 5; ++c) out[c] = z[c];
        for (int j = 0; j < 2; ++j) {
            float a = fc_b[j];
            for (int k = 0; k < 5; ++k) a += z2[k] * fc_W[k * 2 + j];
            out[5 + j] = 1.0f / (1.0f + expf(-a));
        }
    }
}

// ---------------------------------------------------------------------------
extern "C" void kernel_launch(void* const* d_in, const int* in_sizes, int n_in,
                              void* d_out, int out_size, void* d_ws, size_t ws_size,
                              hipStream_t stream) {
    const float* x     = (const float*)d_in[0];
    const float* adj   = (const float*)d_in[1];
    const float* W1    = (const float*)d_in[2];
    const float* b1    = (const float*)d_in[3];
    const float* W2    = (const float*)d_in[4];
    const float* b2    = (const float*)d_in[5];
    const float* fc1_W = (const float*)d_in[6];
    const float* fc1_b = (const float*)d_in[7];
    const float* fc_W  = (const float*)d_in[8];
    const float* fc_b  = (const float*)d_in[9];
    float* out = (float*)d_out;

    // Workspace carve-up (all offsets 16B-aligned).
    char* ws = (char*)d_ws;
    size_t off = 0;
    _Float16* adj16 = (_Float16*)(ws + off); off += (size_t)N * N * sizeof(_Float16); // 128 MiB
    float*    dinv  = (float*)   (ws + off); off += (size_t)N * sizeof(float);
    _Float16* s1T   = (_Float16*)(ws + off); off += (size_t)16 * N * sizeof(_Float16);
    _Float16* s2T   = (_Float16*)(ws + off); off += (size_t)16 * N * sizeof(_Float16);
    float*    hT    = (float*)   (ws + off); off += (size_t)16 * N * sizeof(float);
    float*    colsum= (float*)   (ws + off); off += 16 * sizeof(float);
    (void)ws_size; (void)in_sizes; (void)n_in; (void)out_size;

    // 1) degree + f32->f16 convert of adj (NT reads; f16 copy stays in L2)
    deg_convert_kernel<<<N, 256, 0, stream>>>(adj, adj16, dinv);
    // 2) s1 = dinv * (x @ W1), transposed f16, padded to 16 cols
    s1_kernel<<<(N * 16) / 256, 256, 0, stream>>>(x, W1, dinv, s1T);
    // 3) layer-1 WMMA pass: h = relu(dinv*(adj@s1 + s1) + b1)
    gcn_pass_kernel<1><<<N / 16, 256, 0, stream>>>(adj16, s1T, dinv, b1, hT, colsum);
    // 4) s2 = dinv * (h @ W2), transposed f16, padded; zeroes colsum
    s2_kernel<<<(N * 16) / 256, 256, 0, stream>>>(hT, W2, dinv, s2T, colsum);
    // 5) layer-2 WMMA pass: colsum += dinv*(adj@s2 + s2) + b2
    gcn_pass_kernel<2><<<N / 16, 256, 0, stream>>>(adj16, s2T, dinv, b2, hT, colsum);
    // 6) head: z / y -> d_out (7 floats)
    finalize_kernel<<<1, 32, 0, stream>>>(colsum, fc1_W, fc1_b, fc_W, fc_b, out);
}
